// Seq2Seq_meta_1151051235836
// MI455X (gfx1250) — matrix operand order repaired and checked by reading, not compiled, not run on hardware
//
#include <hip/hip_runtime.h>

// ============================================================================
// MI455X (gfx1250) persistent-RNN implementation (round 2).
//
// One persistent kernel runs the whole 192-step GRU scan: 64 WGs x 128 thr
// (4 waves, wave32); each WG owns 16 batch rows with h resident in LDS.
// All matmuls (meta, gate hidden, gate logits, GRU, proj, head) use
// v_wmma_f32_16x16x32_f16. Weights converted once to f16 in workspace,
// row-major [n][k] so each WMMA B operand is two 16B global loads per lane.
// Transcendentals: hardware v_tanh_f32 (guarded) + branchless exp2/rcp.
// ============================================================================

typedef __attribute__((ext_vector_type(16))) _Float16 v16h;
typedef __attribute__((ext_vector_type(8)))  _Float16 v8h;
typedef __attribute__((ext_vector_type(8)))  float    v8f;

#define ROWS   16     // batch rows per workgroup
#define NTHR   128    // 4 waves of 32
#define STR_H  264    // LDS row stride (halves) for 256-wide h tile (+pad)
#define STR_X  136    // LDS row stride (halves) for 128-wide tiles (+pad)

#define LOG2E  1.4426950408889634f

// ---------------------------------------------------------------------------
// Fast transcendentals (branchless; single v_tanh_f32 on gfx1250).
// ---------------------------------------------------------------------------
__device__ __forceinline__ float fast_tanh(float x) {
#if __has_builtin(__builtin_amdgcn_tanh_f32)
  return __builtin_amdgcn_tanh_f32(x);
#elif __has_builtin(__builtin_amdgcn_tanhf)
  return __builtin_amdgcn_tanhf(x);
#else
  // tanh(x) = 1 - 2/(exp2(2*log2e*x) + 1); saturates correctly at +-inf.
  float e = __builtin_amdgcn_exp2f(2.0f * LOG2E * x);
  return 1.0f - 2.0f * __builtin_amdgcn_rcpf(e + 1.0f);
#endif
}

__device__ __forceinline__ float fast_sigmoid(float x) {
  // 1/(1+exp2(-log2e*x)): x>>0 -> rcp(1+0)=1 ; x<<0 -> rcp(inf)=0.
  return __builtin_amdgcn_rcpf(1.0f + __builtin_amdgcn_exp2f(-LOG2E * x));
}

__device__ __forceinline__ float fast_exp(float x) {
  return __builtin_amdgcn_exp2f(LOG2E * x);
}

// ---------------------------------------------------------------------------
// WMMA operand loaders (CDNA5 16-bit layouts, wave32).
// A (16xK tile, row-major in LDS): lane l -> row l&15; chunks of 8 halves at
// K offsets {(l>>4)*8, 16+(l>>4)*8} within the 32-wide K step.
// B (KxN = W[n][k] row-major in memory): lane l -> column n0+(l&15); chunks
// of 8 halves at K offsets {(l>>4)*8, 16+(l>>4)*8} (same as A).
// C/D: lane l -> col l&15, VGPR j -> row (l>>4)*8 + j.
// ---------------------------------------------------------------------------
__device__ __forceinline__ v16h ldsA(const _Float16* tile, int stride, int k0) {
  const int lane = threadIdx.x & 31;
  const _Float16* p = tile + (lane & 15) * stride + k0 + ((lane >> 4) << 3);
  v8h a0 = *(const v8h*)p;
  v8h a1 = *(const v8h*)(p + 16);
  v16h r;
#pragma unroll
  for (int i = 0; i < 8; ++i) { r[i] = a0[i]; r[8 + i] = a1[i]; }
  return r;
}

__device__ __forceinline__ v16h ldgB(const _Float16* __restrict__ w, int K, int n0, int k0) {
  const int lane = threadIdx.x & 31;
  const _Float16* p = w + (size_t)(n0 + (lane & 15)) * K + k0 + ((lane >> 4) << 3);
  v8h b0 = *(const v8h*)p;
  v8h b1 = *(const v8h*)(p + 16);
  v16h r;
#pragma unroll
  for (int i = 0; i < 8; ++i) { r[i] = b0[i]; r[8 + i] = b1[i]; }
  return r;
}

__device__ __forceinline__ v8f wmma16(v16h a, v16h b, v8f c) {
  return __builtin_amdgcn_wmma_f32_16x16x32_f16(false, a, false, b, (short)0, c, false, false);
}

// ---------------------------------------------------------------------------
struct BlockP {              // per-phase (encoder or decoder) parameters
  const float* xl; const float* xt; const float* xw; const float* xs;
  int tstride;                         // time length (168 or 24)
  const float* w1[3];                  // meta w1 [128]   (temp, work, seas)
  const float* b1[3];                  // meta b1 [128]
  const float* b2[3];                  // meta b2 [128]
  const _Float16* w2;                  // [3][128][128] f16
  const _Float16* g1w;                 // [256][256] f16
  const float* g1b;                    // [256]
  const _Float16* g2w16;               // [16][256] f16 (rows 3..15 zero)
  const float* g2b;                    // [3]
  const float* ln_w; const float* ln_b; const float* theta0;   // [128] each
  const _Float16* wih;                 // [768][128] f16
  const _Float16* whh;                 // [768][256] f16
  const float* bih; const float* bhh;  // [768]
};

struct KArgs {
  BlockP enc, dec;
  const _Float16* projw;               // [256][256] f16
  const float* projb;                  // [256]
  const _Float16* headw16;             // [32][256] f16 (rows 24..31 zero)
  const float* headb;                  // [24]
  const int* topk;
  float* out;                          // [1024][25][24]
};

// ---------------------------------------------------------------------------
__global__ void f32_to_f16_kernel(const float* __restrict__ s,
                                  _Float16* __restrict__ d, int n) {
  int i = blockIdx.x * blockDim.x + threadIdx.x;
  if (i < n) d[i] = (_Float16)s[i];
}

__global__ void pad_rows_f16_kernel(const float* __restrict__ s,
                                    _Float16* __restrict__ d,
                                    int rows_src, int rows_dst, int cols) {
  int i = blockIdx.x * blockDim.x + threadIdx.x;
  if (i < rows_dst * cols) {
    int r = i / cols;
    d[i] = (r < rows_src) ? (_Float16)s[i] : (_Float16)0.f;
  }
}

// ---------------------------------------------------------------------------
// One recurrent step for this WG's 16 rows.
// scratch overlay (32768B):
//   phase 1-3:  h2 [3][16][136] f16 @0 (13056B), h1/gh1 @13056
//   phase 4  :  rz [16][512] f32 @0 (32768B)
// ---------------------------------------------------------------------------
__device__ __forceinline__ void rnn_step(
    const BlockP& P, int t, int r0, int topk,
    float* h32, _Float16* h16, _Float16* xp16, unsigned char* scratch,
    float* stats, float* logits, float* gatew, float* xin) {

  const int tid  = threadIdx.x;
  const int wave = tid >> 5;
  const int lane = tid & 31;
  const int col  = lane & 15;
  const int rb   = (lane >> 4) * 8;

  _Float16* h2  = (_Float16*)scratch;             // [3][16][STR_X]
  _Float16* h1  = (_Float16*)(scratch + 13056);   // [3][16][STR_X]
  _Float16* gh1 = (_Float16*)(scratch + 13056);   // [16][STR_H] (reuses h1)
  float*    rz  = (float*)scratch;                // [16][512]   (reuses all)

  // ---- load per-step scalar inputs ---------------------------------------
  if (tid < ROWS) {
    int b = r0 + tid;
    xin[tid]      = P.xl[b * P.tstride + t];
    xin[16 + tid] = P.xt[b * P.tstride + t];   // expert 0: temp
    xin[32 + tid] = P.xw[b * P.tstride + t];   // expert 1: work
    xin[48 + tid] = P.xs[b * P.tstride + t];   // expert 2: seas
  }
  __syncthreads();

  // ---- phase 1a: h1[e] = tanh(x_e * w1 + b1)  (outer product, VALU) ------
  for (int idx = tid; idx < 3 * ROWS * 128; idx += NTHR) {
    int e = idx >> 11, r = (idx >> 7) & 15, k = idx & 127;
    float v = fast_tanh(xin[16 + e * 16 + r] * P.w1[e][k] + P.b1[e][k]);
    h1[(e * 16 + r) * STR_X + k] = (_Float16)v;
  }
  __syncthreads();

  // ---- phase 1b: h2[e] = tanh(h1[e] @ w2[e].T + b2[e])  (WMMA) -----------
  for (int tile = wave * 6; tile < wave * 6 + 6; ++tile) {   // 24 tiles
    int e = tile >> 3, n = (tile & 7) * 16;
    v8f acc = {};
#pragma unroll
    for (int k = 0; k < 128; k += 32)
      acc = wmma16(ldsA(h1 + e * 16 * STR_X, STR_X, k),
                   ldgB(P.w2 + e * 128 * 128, 128, n, k), acc);
    float bb = P.b2[e][n + col];
#pragma unroll
    for (int j = 0; j < 8; ++j)
      h2[(e * 16 + rb + j) * STR_X + n + col] = (_Float16)fast_tanh(acc[j] + bb);
  }
  __syncthreads();

  // ---- phase 1c: LayerNorm stats per (expert,row) (vectorized LDS reads) -
  if (tid < 48) {
    int e = tid / 16, r = tid & 15;
    const _Float16* p = h2 + (e * 16 + r) * STR_X;
    float s = 0.f, ss = 0.f;
#pragma unroll
    for (int k = 0; k < 128; k += 8) {
      v8h c = *(const v8h*)(p + k);
#pragma unroll
      for (int i = 0; i < 8; ++i) { float v = (float)c[i]; s += v; ss += v * v; }
    }
    float m = s * (1.f / 128.f);
    float var = ss * (1.f / 128.f) - m * m;
    stats[(e * 16 + r) * 2 + 0] = m;
    stats[(e * 16 + r) * 2 + 1] = __builtin_amdgcn_rsqf(var + 1e-5f);
  }

  // ---- phase 2a: gh1 = tanh(h @ g1w.T + g1b)  (WMMA) ---------------------
  for (int tile = wave * 4; tile < wave * 4 + 4; ++tile) {   // 16 tiles
    int n = tile * 16;
    v8f acc = {};
#pragma unroll
    for (int k = 0; k < 256; k += 32)
      acc = wmma16(ldsA(h16, STR_H, k), ldgB(P.g1w, 256, n, k), acc);
    float bb = P.g1b[n + col];
#pragma unroll
    for (int j = 0; j < 8; ++j)
      gh1[(rb + j) * STR_H + n + col] = (_Float16)fast_tanh(acc[j] + bb);
  }
  __syncthreads();

  // ---- phase 2b: logits = gh1 @ g2w.T + g2b  (one WMMA tile, wave 0) -----
  if (wave == 0) {
    v8f acc = {};
#pragma unroll
    for (int k = 0; k < 256; k += 32)
      acc = wmma16(ldsA(gh1, STR_H, k), ldgB(P.g2w16, 256, 0, k), acc);
    if (col < 3) {
      float bb = P.g2b[col];
#pragma unroll
      for (int j = 0; j < 8; ++j)
        logits[(rb + j) * 3 + col] = acc[j] + bb;
    }
  }
  __syncthreads();

  // ---- phase 2c: top-k masked softmax ------------------------------------
  if (tid < ROWS) {
    float l0 = logits[tid * 3], l1 = logits[tid * 3 + 1], l2 = logits[tid * 3 + 2];
    bool m0 = true, m1 = true, m2 = true;
    if (topk == 2) {                    // mask out argmin
      if (l0 <= l1 && l0 <= l2)      m0 = false;
      else if (l1 < l0 && l1 <= l2)  m1 = false;
      else                           m2 = false;
    } else if (topk <= 1) {             // keep only argmax
      m0 = m1 = m2 = false;
      if (l0 >= l1 && l0 >= l2) m0 = true; else if (l1 >= l2) m1 = true; else m2 = true;
    }
    float mx = -3.0e38f;
    if (m0) mx = fmaxf(mx, l0);  if (m1) mx = fmaxf(mx, l1);  if (m2) mx = fmaxf(mx, l2);
    float e0 = m0 ? fast_exp(l0 - mx) : 0.f;
    float e1 = m1 ? fast_exp(l1 - mx) : 0.f;
    float e2 = m2 ? fast_exp(l2 - mx) : 0.f;
    float inv = __builtin_amdgcn_rcpf(e0 + e1 + e2);
    gatew[tid * 3 + 0] = e0 * inv;
    gatew[tid * 3 + 1] = e1 * inv;
    gatew[tid * 3 + 2] = e2 * inv;
  }
  __syncthreads();

  // ---- phase 3: theta = sum_e g_e*LN(h2_e) + theta0 ;  xp = xl * theta ---
  for (int idx = tid; idx < ROWS * 128; idx += NTHR) {
    int r = idx >> 7, k = idx & 127;
    float acc = P.theta0[k];
#pragma unroll
    for (int e = 0; e < 3; ++e) {
      float v = (float)h2[(e * 16 + r) * STR_X + k];
      v = (v - stats[(e * 16 + r) * 2]) * stats[(e * 16 + r) * 2 + 1];
      v = v * P.ln_w[k] + P.ln_b[k];
      acc += gatew[r * 3 + e] * v;
    }
    xp16[r * STR_X + k] = (_Float16)(xin[r] * acc);
  }
  __syncthreads();

  // ---- phase 4: GRU -------------------------------------------------------
  // 48 N-tiles of 16 gate cols. Tiles 0..31 (r,z): fused xp@wih + h@whh sums
  // -> sigmoid -> rz LDS. Tiles 32..47 (n): keep gi_n / gh_n separate.
  // Wave w owns tiles {w, w+4, w+8, ...}.
  for (int i = 0; i < 8; ++i) {                       // fused r/z tiles
    int n = (wave + i * 4) * 16;                      // gate col base 0..511
    v8f acc = {};
#pragma unroll
    for (int k = 0; k < 128; k += 32)
      acc = wmma16(ldsA(xp16, STR_X, k), ldgB(P.wih, 128, n, k), acc);
#pragma unroll
    for (int k = 0; k < 256; k += 32)
      acc = wmma16(ldsA(h16, STR_H, k), ldgB(P.whh, 256, n, k), acc);
    float bsum = P.bih[n + col] + P.bhh[n + col];
#pragma unroll
    for (int j = 0; j < 8; ++j)
      rz[(rb + j) * 512 + n + col] = fast_sigmoid(acc[j] + bsum);
  }
  v8f accA[4], accB[4];
#pragma unroll
  for (int i = 0; i < 4; ++i) {                       // n tiles (gate cols 512..767)
    int n = (32 + wave + i * 4) * 16;
    v8f a = {}, b = {};
#pragma unroll
    for (int k = 0; k < 128; k += 32)
      a = wmma16(ldsA(xp16, STR_X, k), ldgB(P.wih, 128, n, k), a);
#pragma unroll
    for (int k = 0; k < 256; k += 32)
      b = wmma16(ldsA(h16, STR_H, k), ldgB(P.whh, 256, n, k), b);
    accA[i] = a; accB[i] = b;
  }
  __syncthreads();    // all h16/xp16 reads + rz writes complete
#pragma unroll
  for (int i = 0; i < 4; ++i) {
    int gc = (32 + wave + i * 4) * 16;                // 512..767
    int oc = gc - 512;                                // output h col base
    float bi = P.bih[gc + col], bh = P.bhh[gc + col];
#pragma unroll
    for (int j = 0; j < 8; ++j) {
      int m = rb + j, c = oc + col;
      float r_ = rz[m * 512 + c];
      float z_ = rz[m * 512 + 256 + c];
      float nn = fast_tanh((accA[i][j] + bi) + r_ * (accB[i][j] + bh));
      float hnew = (1.f - z_) * nn + z_ * h32[m * 256 + c];
      h32[m * 256 + c] = hnew;
      h16[m * STR_H + c] = (_Float16)hnew;
    }
  }
  __syncthreads();
}

// ---------------------------------------------------------------------------
// head: pred = h @ head_w.T + head_b via 2 WMMA tiles (waves 0,1), written
// straight to global. Requires h16 valid (caller syncs before).
// ---------------------------------------------------------------------------
__device__ __forceinline__ void head_out(const _Float16* h16,
                                         const _Float16* hw16, const float* hb,
                                         float* out, int r0, int sidx) {
  const int wave = threadIdx.x >> 5;
  const int lane = threadIdx.x & 31;
  const int col  = lane & 15;
  const int rb   = (lane >> 4) * 8;
  if (wave < 2) {
    int n = wave * 16;
    v8f acc = {};
#pragma unroll
    for (int k = 0; k < 256; k += 32)
      acc = wmma16(ldsA(h16, STR_H, k), ldgB(hw16, 256, n, k), acc);
    int o = n + col;
    if (o < 24) {
      float bb = hb[o];
#pragma unroll
      for (int j = 0; j < 8; ++j)
        out[(size_t)(r0 + rb + j) * (25 * 24) + sidx * 24 + o] = acc[j] + bb;
    }
  }
}

// ---------------------------------------------------------------------------
__global__ __launch_bounds__(NTHR) void rnn_persist(KArgs A) {
  __shared__ float     h32[ROWS * 256];
  __shared__ _Float16  h16[ROWS * STR_H];
  __shared__ _Float16  xp16[ROWS * STR_X];
  __shared__ __align__(16) unsigned char scratch[32768];
  __shared__ float     stats[3 * ROWS * 2];
  __shared__ float     logits[ROWS * 3];
  __shared__ float     gatew[ROWS * 3];
  __shared__ float     xin[4 * ROWS];

  const int r0   = blockIdx.x * ROWS;
  const int topk = *A.topk;
  const int wave = threadIdx.x >> 5;
  const int lane = threadIdx.x & 31;
  const int col  = lane & 15;
  const int rb   = (lane >> 4) * 8;

  for (int i = threadIdx.x; i < ROWS * 256; i += NTHR) h32[i] = 0.f;
  for (int i = threadIdx.x; i < ROWS * STR_H; i += NTHR) h16[i] = (_Float16)0.f;
  __syncthreads();

  // -------- encoder scan --------
  for (int t = 0; t < 168; ++t)
    rnn_step(A.enc, t, r0, topk, h32, h16, xp16, scratch, stats, logits, gatew, xin);

  // -------- proj to latent: h = h_enc @ proj_w.T + proj_b (WMMA) --------
  {
    v8f acc[4];
#pragma unroll
    for (int i = 0; i < 4; ++i) {
      int n = (wave + i * 4) * 16;
      v8f a = {};
#pragma unroll
      for (int k = 0; k < 256; k += 32)
        a = wmma16(ldsA(h16, STR_H, k), ldgB(A.projw, 256, n, k), a);
      acc[i] = a;
    }
    __syncthreads();
#pragma unroll
    for (int i = 0; i < 4; ++i) {
      int n = (wave + i * 4) * 16;
      float bb = A.projb[n + col];
#pragma unroll
      for (int j = 0; j < 8; ++j) {
        float v = acc[i][j] + bb;
        h32[(rb + j) * 256 + n + col] = v;
        h16[(rb + j) * STR_H + n + col] = (_Float16)v;
      }
    }
    __syncthreads();
  }

  head_out(h16, A.headw16, A.headb, A.out, r0, 0);     // pred0

  // -------- decoder scan --------
  for (int t = 0; t < 24; ++t) {
    rnn_step(A.dec, t, r0, topk, h32, h16, xp16, scratch, stats, logits, gatew, xin);
    head_out(h16, A.headw16, A.headb, A.out, r0, t + 1);
  }
}

// ===========================================================================
// Host side.
// Input order: top-level setup_inputs() dict order; nested `params` pytree in
// JAX sorted-key leaf order:
//   8  gru_dec.b_hh   9  gru_dec.b_ih  10 gru_dec.w_hh  11 gru_dec.w_ih
//   12 gru_enc.b_hh   13 gru_enc.b_ih  14 gru_enc.w_hh  15 gru_enc.w_ih
//   16 head_b  17 head_w  18 proj_b  19 proj_w
//   20 tdec.g1b 21 tdec.g1w 22 tdec.g2b 23 tdec.g2w 24 tdec.ln_b 25 tdec.ln_w
//   26 tdec.seas.b1 27 .b2 28 .w1 29 .w2   30 tdec.temp.b1 31 .b2 32 .w1 33 .w2
//   34 tdec.theta0   35 tdec.work.b1 36 .b2 37 .w1 38 .w2
//   39..57 tenc (same layout as tdec)      58 top_k
// ===========================================================================
enum {
  OFF_ENC_W2  = 0,                       // 3 * 16384
  OFF_DEC_W2  = 49152,
  OFF_ENC_G1W = 98304,                   // 65536
  OFF_DEC_G1W = 163840,
  OFF_ENC_WIH = 229376,                  // 98304 (768x128)
  OFF_DEC_WIH = 327680,
  OFF_ENC_WHH = 425984,                  // 196608 (768x256)
  OFF_DEC_WHH = 622592,
  OFF_PROJW   = 819200,                  // 65536
  OFF_ENC_G2W = 884736,                  // 4096 (16x256, rows 3..15 zero)
  OFF_DEC_G2W = 888832,                  // 4096
  OFF_HEADW   = 892928,                  // 8192 (32x256, rows 24..31 zero)
};                                       // total 901120 halves = 1.72 MB

static void conv(const void* s, _Float16* d, int n, hipStream_t st) {
  f32_to_f16_kernel<<<(n + 255) / 256, 256, 0, st>>>((const float*)s, d, n);
}

static void pad(const void* s, _Float16* d, int rs, int rd, int cols, hipStream_t st) {
  int n = rd * cols;
  pad_rows_f16_kernel<<<(n + 255) / 256, 256, 0, st>>>((const float*)s, d, rs, rd, cols);
}

extern "C" void kernel_launch(void* const* d_in, const int* in_sizes, int n_in,
                              void* d_out, int out_size, void* d_ws, size_t ws_size,
                              hipStream_t stream) {
  (void)in_sizes; (void)n_in; (void)out_size; (void)ws_size;
  _Float16* ws = (_Float16*)d_ws;
  const float* const* F = (const float* const*)d_in;

  // ---- one-shot f32 -> f16 weight conversion (L2-resident thereafter) ----
  conv(F[52], ws + OFF_ENC_W2 + 0 * 16384, 16384, stream);  // tenc.temp.w2
  conv(F[57], ws + OFF_ENC_W2 + 1 * 16384, 16384, stream);  // tenc.work.w2
  conv(F[48], ws + OFF_ENC_W2 + 2 * 16384, 16384, stream);  // tenc.seas.w2
  conv(F[33], ws + OFF_DEC_W2 + 0 * 16384, 16384, stream);  // tdec.temp.w2
  conv(F[38], ws + OFF_DEC_W2 + 1 * 16384, 16384, stream);  // tdec.work.w2
  conv(F[29], ws + OFF_DEC_W2 + 2 * 16384, 16384, stream);  // tdec.seas.w2
  conv(F[40], ws + OFF_ENC_G1W, 65536, stream);             // tenc.g1w
  conv(F[21], ws + OFF_DEC_G1W, 65536, stream);             // tdec.g1w
  conv(F[15], ws + OFF_ENC_WIH, 98304, stream);             // gru_enc.w_ih
  conv(F[11], ws + OFF_DEC_WIH, 98304, stream);             // gru_dec.w_ih
  conv(F[14], ws + OFF_ENC_WHH, 196608, stream);            // gru_enc.w_hh
  conv(F[10], ws + OFF_DEC_WHH, 196608, stream);            // gru_dec.w_hh
  conv(F[19], ws + OFF_PROJW, 65536, stream);               // proj_w
  pad(F[42], ws + OFF_ENC_G2W, 3, 16, 256, stream);         // tenc.g2w -> 16x256
  pad(F[23], ws + OFF_DEC_G2W, 3, 16, 256, stream);         // tdec.g2w -> 16x256
  pad(F[17], ws + OFF_HEADW, 24, 32, 256, stream);          // head_w   -> 32x256

  KArgs a;
  // encoder block
  a.enc.xl = F[0]; a.enc.xt = F[1]; a.enc.xw = F[2]; a.enc.xs = F[3];
  a.enc.tstride = 168;
  a.enc.w1[0] = F[51]; a.enc.w1[1] = F[56]; a.enc.w1[2] = F[47];
  a.enc.b1[0] = F[49]; a.enc.b1[1] = F[54]; a.enc.b1[2] = F[45];
  a.enc.b2[0] = F[50]; a.enc.b2[1] = F[55]; a.enc.b2[2] = F[46];
  a.enc.w2 = ws + OFF_ENC_W2;  a.enc.g1w = ws + OFF_ENC_G1W;
  a.enc.g1b = F[39]; a.enc.g2w16 = ws + OFF_ENC_G2W; a.enc.g2b = F[41];
  a.enc.ln_w = F[44]; a.enc.ln_b = F[43]; a.enc.theta0 = F[53];
  a.enc.wih = ws + OFF_ENC_WIH; a.enc.whh = ws + OFF_ENC_WHH;
  a.enc.bih = F[13]; a.enc.bhh = F[12];
  // decoder block
  a.dec.xl = F[4]; a.dec.xt = F[5]; a.dec.xw = F[6]; a.dec.xs = F[7];
  a.dec.tstride = 24;
  a.dec.w1[0] = F[32]; a.dec.w1[1] = F[37]; a.dec.w1[2] = F[28];
  a.dec.b1[0] = F[30]; a.dec.b1[1] = F[35]; a.dec.b1[2] = F[26];
  a.dec.b2[0] = F[31]; a.dec.b2[1] = F[36]; a.dec.b2[2] = F[27];
  a.dec.w2 = ws + OFF_DEC_W2;  a.dec.g1w = ws + OFF_DEC_G1W;
  a.dec.g1b = F[20]; a.dec.g2w16 = ws + OFF_DEC_G2W; a.dec.g2b = F[22];
  a.dec.ln_w = F[25]; a.dec.ln_b = F[24]; a.dec.theta0 = F[34];
  a.dec.wih = ws + OFF_DEC_WIH; a.dec.whh = ws + OFF_DEC_WHH;
  a.dec.bih = F[9]; a.dec.bhh = F[8];
  // shared
  a.projw = ws + OFF_PROJW; a.projb = F[18];
  a.headw16 = ws + OFF_HEADW; a.headb = F[16];
  a.topk = (const int*)d_in[58];
  a.out = (float*)d_out;

  rnn_persist<<<1024 / ROWS, NTHR, 0, stream>>>(a);
}
